// unit_gcn_70076686401609
// MI455X (gfx1250) — matrix operand order repaired and checked by reading, not compiled, not run on hardware
//
#include <hip/hip_runtime.h>
#include <hip/hip_bf16.h>
#include <math.h>

// Problem constants (from reference setup_inputs)
#define N_   64
#define C_   64
#define T_   300
#define V_   21
#define S_   3
#define IC_  16
#define OC_  64
#define TV_  (T_*V_)   // 6300
#define VV_  (V_*V_)   // 441
#define NEDGE 231      // V*(V+1)/2 upper-tri incl diagonal

// Workspace layout (float offsets)
#define AEFF_OFF   0          // S_*VV_ = 1323
#define STATS_OFF  1536       // 128 (sum, sumsq per channel)
#define SCORES_OFF 2048       // N_*S_*VV_ = 84672 (scores, then att in-place)
#define Y_OFF      87040      // N_*OC_*T_*V_ = 25804800

typedef __attribute__((ext_vector_type(16))) __bf16 v16bf;
typedef __attribute__((ext_vector_type(8)))  __bf16 v8bf;
typedef __attribute__((ext_vector_type(8)))  float  v8f;

__device__ __forceinline__ v8f wmma_bf16(v16bf a, v16bf b, v8f c) {
  // D = A(16x32 bf16) * B(32x16 bf16) + C(16x16 f32)
  return __builtin_amdgcn_wmma_f32_16x16x32_bf16(false, a, false, b, (short)0, c, false, false);
}

__device__ __forceinline__ v16bf cat8(v8bf a, v8bf b) {
  return __builtin_shufflevector(a, b, 0,1,2,3,4,5,6,7,8,9,10,11,12,13,14,15);
}

// A operand (16x32 bf16) from ROW-MAJOR LDS tile base[m*stride + k].
// Per-lane data is two contiguous 8-half runs: k = 8h + e (e<8), k = 16+8h+(e-8).
// => exactly 2x ds_load_b128 per lane.
__device__ __forceinline__ v16bf ldA(const __bf16* base, int stride, int lane) {
  const int m = lane & 15, h = (lane >> 4) & 1;
  const __bf16* p = base + m * stride + 8 * h;
  v8bf lo = *(const v8bf*)(p);
  v8bf hi = *(const v8bf*)(p + 16);
  return cat8(lo, hi);
}

// B operand (32x16 bf16) from TRANSPOSED LDS tile baseT[n*stride + k].
// Per-lane data is two contiguous 8-half runs: k = 8h + e (e<8), k = 16+8h+(e-8).
// => exactly 2x ds_load_b128 per lane.
__device__ __forceinline__ v16bf ldBT(const __bf16* baseT, int stride, int lane) {
  const int n = lane & 15, h = (lane >> 4) & 1;
  const __bf16* p = baseT + n * stride + 8 * h;
  v8bf lo = *(const v8bf*)(p);
  v8bf hi = *(const v8bf*)(p + 16);
  return cat8(lo, hi);
}

// ---------------------------------------------------------------------------
// K0: maximum-spanning-tree (naive Kruskal, faithful to reference) + A_eff,
//     and zero BN stats (re-done every launch => graph-replay safe).
// ---------------------------------------------------------------------------
__global__ void k0_mst(const float* __restrict__ A, const float* __restrict__ PA,
                       float* __restrict__ ws) {
  const int tid = threadIdx.x;
  if (tid < 128) ws[STATS_OFF + tid] = 0.f;
  if (tid < S_) {
    const int s = tid;
    float* aeff = ws + AEFF_OFF + s*VV_;
    for (int k = 0; k < VV_; ++k) aeff[k] = A[s*VV_ + k];
    float ew[NEDGE]; int e0[NEDGE], e1[NEDGE];
    int ne = 0;
    for (int i = 0; i < V_; ++i)
      for (int j = i; j < V_; ++j) {
        e0[ne] = i; e1[ne] = j; ew[ne] = PA[s*VV_ + i*V_ + j]; ++ne;
      }
    for (int a = 0; a < NEDGE - 1; ++a) {
      int best = a;
      for (int b = a + 1; b < NEDGE; ++b) if (ew[b] > ew[best]) best = b;
      if (best != a) {
        float tw = ew[a]; ew[a] = ew[best]; ew[best] = tw;
        int t0 = e0[a]; e0[a] = e0[best]; e0[best] = t0;
        int t1 = e1[a]; e1[a] = e1[best]; e1[best] = t1;
      }
    }
    int grp[V_];
    for (int v = 0; v < V_; ++v) grp[v] = v;
    for (int e = 0; e < NEDGE; ++e) {
      const int gm = grp[e0[e]], gn = grp[e1[e]];
      if (gm != gn) {
        aeff[e0[e]*V_ + e1[e]] += ew[e];
        for (int v = 0; v < V_; ++v) if (grp[v] == gn) grp[v] = gm;
      }
    }
  }
}

// ---------------------------------------------------------------------------
// K1: scores[n,s,v,w] = (1/(IC*T)) * sum_{i,t} fa[i,t,v]*fb[i,t,w]
//   per t:  fa/fb tiles via WMMA (Wa/Wb as A, x^T staged for B),
//           then scores += fa^T * fb via WMMA (K=16 real + 16 zero-pad),
//           f32 accumulators live in WMMA C registers across the T loop.
// Block = 128 threads (4 waves), one block per (n,s).
// ---------------------------------------------------------------------------
__global__ __launch_bounds__(128)
void k1_scores(const float* __restrict__ x,
               const float* __restrict__ Wa, const float* __restrict__ ba,
               const float* __restrict__ Wb, const float* __restrict__ bb,
               float* __restrict__ ws) {
  const int n = blockIdx.x / S_, s = blockIdx.x % S_;
  const int tid = threadIdx.x, wave = tid >> 5, lane = tid & 31;
  const int h = (lane >> 4) & 1;

  __shared__ __attribute__((aligned(16))) __bf16 sWa[IC_*C_];      // 2 KB  [i][c]
  __shared__ __attribute__((aligned(16))) __bf16 sWb[IC_*C_];      // 2 KB  [i][c]
  __shared__ __attribute__((aligned(16))) __bf16 sX[4][32*C_];     // 16 KB [v][c] (transposed, v padded/zeroed)
  __shared__ __attribute__((aligned(16))) __bf16 sFaT[4][32*32];   // 8 KB  [v][i] (k>=16 stays zero)
  __shared__ __attribute__((aligned(16))) __bf16 sFbT[4][32*32];   // 8 KB  [v][i]
  __shared__ __attribute__((aligned(16))) float  sScore[32*32];    // 4 KB

  for (int idx = tid; idx < IC_*C_; idx += 128) {
    sWa[idx] = (__bf16)Wa[s*IC_*C_ + idx];
    sWb[idx] = (__bf16)Wb[s*IC_*C_ + idx];
  }
  {
    const v8bf z = {};
    for (int idx = tid; idx < 4*32*32/8; idx += 128) {        // zero fa/fb (k-pad)
      ((v8bf*)sFaT)[idx] = z;
      ((v8bf*)sFbT)[idx] = z;
    }
    for (int idx = tid; idx < 32*32; idx += 128) sScore[idx] = 0.f;
  }
  __syncthreads();

  // Wa/Wb as WMMA A operands (rows i, K = c; two 32-wide K chunks)
  v16bf aWa0 = ldA(sWa,      C_, lane);
  v16bf aWa1 = ldA(sWa + 32, C_, lane);
  v16bf aWb0 = ldA(sWb,      C_, lane);
  v16bf aWb1 = ldA(sWb + 32, C_, lane);

  float bA[8], bB[8];
#pragma unroll
  for (int r = 0; r < 8; ++r) {
    bA[r] = ba[s*IC_ + r + 8*h];
    bB[r] = bb[s*IC_ + r + 8*h];
  }

  v8f sc[2][2] = {};
  __bf16* xw  = sX[wave];
  __bf16* faT = sFaT[wave];
  __bf16* fbT = sFbT[wave];

  for (int t = wave; t < T_; t += 4) {
    // stage x[n,:,t,:] transposed -> [v][c], one lane per row v (zeros v>=21)
    {
      const int v = lane;
      __bf16* row = xw + v * C_;
#pragma unroll
      for (int g = 0; g < 8; ++g) {
        v8bf pk;
#pragma unroll
        for (int e = 0; e < 8; ++e) {
          const int c = g*8 + e;
          const float val = (v < V_) ? x[((n*C_ + c)*T_ + t)*V_ + v] : 0.f;
          pk[e] = (__bf16)val;
        }
        *(v8bf*)(row + g*8) = pk;
      }
    }
    // fa/fb tiles: 2 v-tiles x 2 K-chunks, B shared between fa and fb
#pragma unroll
    for (int vt = 0; vt < 2; ++vt) {
      v16bf b0 = ldBT(xw + vt*16*C_,      C_, lane);   // c = 0..31
      v16bf b1 = ldBT(xw + vt*16*C_ + 32, C_, lane);   // c = 32..63
      v8f fa = {}; fa = wmma_bf16(aWa0, b0, fa); fa = wmma_bf16(aWa1, b1, fa);
      v8f fb = {}; fb = wmma_bf16(aWb0, b0, fb); fb = wmma_bf16(aWb1, b1, fb);
      const int v = vt*16 + (lane & 15);
      v8bf pa, pb;
#pragma unroll
      for (int r = 0; r < 8; ++r) {
        pa[r] = (__bf16)(fa[r] + bA[r]);               // row i = r + 8h
        pb[r] = (__bf16)(fb[r] + bB[r]);
      }
      *(v8bf*)(faT + v*32 + 8*h) = pa;                 // [v][i], i = 8h..8h+7
      *(v8bf*)(fbT + v*32 + 8*h) = pb;
    }
    // scores += fa^T(v x i) * fb(i x w), K = 32 (upper 16 are zeros)
#pragma unroll
    for (int vt2 = 0; vt2 < 2; ++vt2) {
      v16bf aS = ldA(faT + vt2*16*32, 32, lane);
#pragma unroll
      for (int wt2 = 0; wt2 < 2; ++wt2) {
        v16bf bS = ldBT(fbT + wt2*16*32, 32, lane);
        sc[vt2][wt2] = wmma_bf16(aS, bS, sc[vt2][wt2]);
      }
    }
  }
  // cross-wave reduction via LDS float atomics
#pragma unroll
  for (int vt2 = 0; vt2 < 2; ++vt2)
#pragma unroll
    for (int wt2 = 0; wt2 < 2; ++wt2)
#pragma unroll
      for (int r = 0; r < 8; ++r) {
        const int v = vt2*16 + r + 8*h;
        const int w = wt2*16 + (lane & 15);
        atomicAdd(&sScore[v*32 + w], sc[vt2][wt2][r]);
      }
  __syncthreads();
  float* scores = ws + SCORES_OFF + (n*S_ + s)*VV_;
  for (int p = tid; p < VV_; p += 128)
    scores[p] = sScore[(p / V_)*32 + (p % V_)] * (1.f / (float)(IC_*T_));
}

// ---------------------------------------------------------------------------
// K2: att = softmax(scores, axis=-2 (over v)) + A_eff   (in place, tiny)
// ---------------------------------------------------------------------------
__global__ void k2_softmax(float* __restrict__ ws) {
  const int n = blockIdx.x / S_, s = blockIdx.x % S_;
  const int w = threadIdx.x;
  if (w >= V_) return;
  float* scp = ws + SCORES_OFF + (n*S_ + s)*VV_;
  const float* aeff = ws + AEFF_OFF + s*VV_;
  float col[V_];
  float mx = -3.4e38f;
  for (int v = 0; v < V_; ++v) { col[v] = scp[v*V_ + w]; mx = fmaxf(mx, col[v]); }
  float sum = 0.f;
  for (int v = 0; v < V_; ++v) { col[v] = expf(col[v] - mx); sum += col[v]; }
  const float inv = 1.f / sum;
  for (int v = 0; v < V_; ++v) scp[v*V_ + w] = col[v]*inv + aeff[v*V_ + w];
}

// ---------------------------------------------------------------------------
// K3: y[n,o,t,w] = sum_s Wd_s @ (x_t @ att_s) + sum_s bd, fused per t.
// Grid (10 t-chunks, N). 8 waves = 4 o/c-tiles x 2 w-tiles.
// All operands 2x ds_load_b128; all spills 1x ds_store_b128.
// ---------------------------------------------------------------------------
__global__ __launch_bounds__(256)
void k3_y(const float* __restrict__ x, const float* __restrict__ Wd,
          const float* __restrict__ bd, float* __restrict__ ws) {
  const int n = blockIdx.y;
  const int tbase = blockIdx.x * 30;
  const int tid = threadIdx.x, wave = tid >> 5, lane = tid & 31;
  const int otile = wave >> 1, wtile = wave & 1;
  const int h = (lane >> 4) & 1;

  __shared__ __attribute__((aligned(16))) __bf16 sWd[S_][OC_*C_];  // 24 KB [o][c]
  __shared__ __attribute__((aligned(16))) __bf16 sAttT[S_][32*32]; // 6 KB  [w][v] transposed
  __shared__ __attribute__((aligned(16))) __bf16 sXt[C_*32];       // 4 KB  [c][v]
  __shared__ __attribute__((aligned(16))) __bf16 sAggT[32*C_];     // 4 KB  [w][c] transposed
  __shared__ float  sBd[OC_];

  for (int idx = tid; idx < S_*OC_*C_; idx += 256)
    sWd[idx / (OC_*C_)][idx % (OC_*C_)] = (__bf16)Wd[idx];
  const float* att = ws + SCORES_OFF + n*S_*VV_;
  for (int idx = tid; idx < S_*32*32; idx += 256) {
    const int s = idx >> 10, r = idx & 1023;
    const int w = r >> 5, v = r & 31;
    const float val = (v < V_ && w < V_) ? att[s*VV_ + v*V_ + w] : 0.f;
    sAttT[s][r] = (__bf16)val;                 // r = w*32 + v
  }
  if (tid < OC_) sBd[tid] = bd[tid] + bd[OC_ + tid] + bd[2*OC_ + tid];
  __syncthreads();

  v16bf aWd[S_][2];
#pragma unroll
  for (int s = 0; s < S_; ++s) {
    aWd[s][0] = ldA(&sWd[s][otile*16*C_],      C_, lane);
    aWd[s][1] = ldA(&sWd[s][otile*16*C_ + 32], C_, lane);
  }
  float* yws = ws + Y_OFF;

  for (int tt = 0; tt < 30; ++tt) {
    const int t = tbase + tt;
    {
      // stage x_t row-major [c][32]: each thread packs 8 halves -> 1 b128 store
      const int c = tid >> 2, v0 = (tid & 3) * 8;
      v8bf pk;
#pragma unroll
      for (int e = 0; e < 8; ++e) {
        const int v = v0 + e;
        const float val = (v < V_) ? x[((n*C_ + c)*T_ + t)*V_ + v] : 0.f;
        pk[e] = (__bf16)val;
      }
      *(v8bf*)&sXt[c*32 + v0] = pk;
    }
    __syncthreads();
    v8f yacc = {};
#pragma unroll
    for (int s = 0; s < S_; ++s) {
      // agg tile: (c = otile block) x (w = wtile block), K = v (padded 32)
      v16bf ax  = ldA(&sXt[otile*16*32], 32, lane);
      v16bf bat = ldBT(&sAttT[s][wtile*16*32], 32, lane);
      v8f agg = {};
      agg = wmma_bf16(ax, bat, agg);
      {
        v8bf pk;
#pragma unroll
        for (int r = 0; r < 8; ++r) pk[r] = (__bf16)agg[r];   // row c = otile*16+r+8h
        const int w = wtile*16 + (lane & 15);
        *(v8bf*)&sAggT[w*C_ + otile*16 + 8*h] = pk;           // [w][c]
      }
      __syncthreads();
      // y tile: (o = otile) x (w = wtile), K = c (2 chunks of 32)
      v16bf b0 = ldBT(&sAggT[wtile*16*C_],      C_, lane);
      v16bf b1 = ldBT(&sAggT[wtile*16*C_ + 32], C_, lane);
      yacc = wmma_bf16(aWd[s][0], b0, yacc);
      yacc = wmma_bf16(aWd[s][1], b1, yacc);
      __syncthreads();
    }
    const int w0 = wtile*16 + (lane & 15);
    if (w0 < V_) {
#pragma unroll
      for (int r = 0; r < 8; ++r) {
        const int o = otile*16 + r + 8*h;
        yws[((n*OC_ + o)*T_ + t)*V_ + w0] = yacc[r] + sBd[o];
      }
    }
    __syncthreads();
  }
}

// ---------------------------------------------------------------------------
// K4: per-channel sum / sumsq over (N,T,V) -> stats (float atomics)
// ---------------------------------------------------------------------------
__global__ __launch_bounds__(256)
void k4_stats(float* __restrict__ ws) {
  const int o = blockIdx.y, chunk = blockIdx.x;
  const int tid = threadIdx.x;
  const float* y = ws + Y_OFF;
  const int per = (N_*TV_) / 16;   // 25200
  float sum = 0.f, sq = 0.f;
  for (int f = chunk*per + tid; f < (chunk + 1)*per; f += 256) {
    const int nn = f / TV_, r = f % TV_;
    const float v = y[(nn*OC_ + o)*TV_ + r];
    sum += v; sq += v*v;
  }
  __shared__ float s1[256], s2[256];
  s1[tid] = sum; s2[tid] = sq;
  __syncthreads();
  for (int st = 128; st > 0; st >>= 1) {
    if (tid < st) { s1[tid] += s1[tid + st]; s2[tid] += s2[tid + st]; }
    __syncthreads();
  }
  if (tid == 0) {
    atomicAdd(ws + STATS_OFF + o,       s1[0]);
    atomicAdd(ws + STATS_OFF + OC_ + o, s2[0]);
  }
}

// ---------------------------------------------------------------------------
// K5: out = relu(gamma*(y-mean)*rsqrt(var+eps) + beta + x)
// ---------------------------------------------------------------------------
__global__ __launch_bounds__(256)
void k5_bn(const float* __restrict__ x, const float* __restrict__ gamma,
           const float* __restrict__ beta, const float* __restrict__ ws,
           float* __restrict__ out) {
  const long long idx = (long long)blockIdx.x * 256 + threadIdx.x;
  const long long total = (long long)N_ * OC_ * TV_;
  if (idx >= total) return;
  const int o = (int)((idx / TV_) % OC_);
  const float* stats = ws + STATS_OFF;
  const float Minv = 1.f / (float)(N_*TV_);
  const float mean = stats[o] * Minv;
  const float var  = stats[OC_ + o] * Minv - mean*mean;
  const float yv   = ws[Y_OFF + idx];
  const float r = gamma[o] * (yv - mean) * rsqrtf(var + 1e-5f) + beta[o] + x[idx];
  out[idx] = fmaxf(r, 0.f);
}

// ---------------------------------------------------------------------------
extern "C" void kernel_launch(void* const* d_in, const int* in_sizes, int n_in,
                              void* d_out, int out_size, void* d_ws, size_t ws_size,
                              hipStream_t stream) {
  (void)in_sizes; (void)n_in; (void)out_size; (void)ws_size;
  const float* x     = (const float*)d_in[0];
  const float* A     = (const float*)d_in[1];
  const float* PA    = (const float*)d_in[2];
  const float* Wa    = (const float*)d_in[3];
  const float* ba    = (const float*)d_in[4];
  const float* Wb    = (const float*)d_in[5];
  const float* bb    = (const float*)d_in[6];
  const float* Wd    = (const float*)d_in[7];
  const float* bd    = (const float*)d_in[8];
  const float* gamma = (const float*)d_in[9];
  const float* beta  = (const float*)d_in[10];
  float* ws  = (float*)d_ws;
  float* out = (float*)d_out;

  hipLaunchKernelGGL(k0_mst,     dim3(1),       dim3(128), 0, stream, A, PA, ws);
  hipLaunchKernelGGL(k1_scores,  dim3(N_*S_),   dim3(128), 0, stream, x, Wa, ba, Wb, bb, ws);
  hipLaunchKernelGGL(k2_softmax, dim3(N_*S_),   dim3(32),  0, stream, ws);
  hipLaunchKernelGGL(k3_y,       dim3(10, N_),  dim3(256), 0, stream, x, Wd, bd, ws);
  hipLaunchKernelGGL(k4_stats,   dim3(16, OC_), dim3(256), 0, stream, ws);
  const long long total = (long long)N_ * OC_ * TV_;
  hipLaunchKernelGGL(k5_bn, dim3((unsigned)((total + 255) / 256)), dim3(256), 0, stream,
                     x, gamma, beta, ws, out);
}
